// LocalWindowAttn_87230785781977
// MI455X (gfx1250) — compile-verified, matrix operand order checked by reading, not run on hardware
//
#include <hip/hip_runtime.h>
#include <hip/hip_bf16.h>

// CDNA5 / gfx1250 wave32 WMMA types
typedef __attribute__((ext_vector_type(16))) _Float16 v16h;
typedef __attribute__((ext_vector_type(8)))  _Float16 v8h;
typedef __attribute__((ext_vector_type(8)))  float    v8f;

// D = A*B + C, 16x16x32 f16 -> f32 accumulate
#define WMMA(a, b, c) \
  __builtin_amdgcn_wmma_f32_16x16x32_f16(false, (a), false, (b), (short)0, (c), false, false)

// Load one 16x32 f16 A (or B^T) fragment for the current lane.
// ISA layout: lanes 0-15 hold K=0..7 (v0-3) and K=16..23 (v4-7),
//             lanes 16-31 hold K=8..15 and K=24..31.
// Caller passes p = row_base + k0 + 8*hi; chunk1 lives at p+16.
__device__ __forceinline__ v16h lwa_load_frag(const _Float16* p) {
  v8h c0 = *(const v8h*)(p);
  v8h c1 = *(const v8h*)(p + 16);
  v16h f;
#pragma unroll
  for (int i = 0; i < 8; ++i) { f[i] = c0[i]; f[i + 8] = c1[i]; }
  return f;
}

// Convert the three fp32 weight matrices to f16 once into workspace.
__global__ void lwa_prep_weights(const float* __restrict__ wqkv,
                                 const float* __restrict__ wmo,
                                 const float* __restrict__ wout,
                                 _Float16* __restrict__ dst) {
  const int i = blockIdx.x * 256 + threadIdx.x;
  const int NQ = 384 * 128, NM = 128 * 128;
  if (i < NQ)                dst[i] = (_Float16)wqkv[i];
  else if (i < NQ + NM)      dst[i] = (_Float16)wmo[i - NQ];
  else if (i < NQ + 2 * NM)  dst[i] = (_Float16)wout[i - NQ - NM];
}

// One workgroup (4 waves) per 7x7 window. B=16, H=W=112, C=128, NH=4, hd=32.
__global__ __launch_bounds__(128) void lwa_win_attn(
    const float* __restrict__ x,
    const float* __restrict__ ln_g,
    const float* __restrict__ ln_b,
    const float* __restrict__ b_qkv,
    const float* __restrict__ b_mo,
    const float* __restrict__ b_out,
    const _Float16* __restrict__ wqkv,   // [384][128] f16
    const _Float16* __restrict__ wmo,    // [128][128] f16
    const _Float16* __restrict__ wout,   // [128][128] f16
    float* __restrict__ out)
{
  __shared__ _Float16 sQKV[64 * 384];   // 48 KB : Q|K|V, rows = tokens (padded to 64)
  __shared__ float    sS[4 * 64 * 64];  // 64 KB : per-head scores; probs alias lower
                                        //         half, V^T reuses part of upper half
  __shared__ _Float16 sO[64 * 128];     // 16 KB : attention output
  __shared__ _Float16 sXn[64 * 128];    // 16 KB : LN(x); reused as MO after QKV GEMM

  const int tid  = threadIdx.x;
  const int lane = tid & 31;
  const int wave = tid >> 5;
  const int l15  = lane & 15;
  const int hi   = lane >> 4;

  const int wg   = blockIdx.x;
  const int bidx = wg >> 8;
  const int rem  = wg & 255;
  const int wh   = rem >> 4, ww = rem & 15;
  const long gbase = (long)bidx * 12544 + (long)(wh * 7) * 112 + (long)(ww * 7);

  // ---- Stage 1: LayerNorm -> sXn (f16), rows 49..63 zeroed ----
  {
    const float4 g4 = *(const float4*)(ln_g + lane * 4);
    const float4 be = *(const float4*)(ln_b + lane * 4);
    for (int t = wave; t < 49; t += 4) {
      const int r = t / 7, c = t - r * 7;
      const float* xp = x + (gbase + r * 112 + c) * 128;
      const float4 v = *(const float4*)(xp + lane * 4);
      float s  = v.x + v.y + v.z + v.w;
      float ss = v.x * v.x + v.y * v.y + v.z * v.z + v.w * v.w;
#pragma unroll
      for (int off = 16; off > 0; off >>= 1) {
        s  += __shfl_xor(s,  off, 32);
        ss += __shfl_xor(ss, off, 32);
      }
      const float mu  = s * (1.0f / 128.0f);
      const float var = ss * (1.0f / 128.0f) - mu * mu;
      const float rs  = rsqrtf(var + 1e-5f);
      _Float16* dst = sXn + t * 128 + lane * 4;
      dst[0] = (_Float16)((v.x - mu) * rs * g4.x + be.x);
      dst[1] = (_Float16)((v.y - mu) * rs * g4.y + be.y);
      dst[2] = (_Float16)((v.z - mu) * rs * g4.z + be.z);
      dst[3] = (_Float16)((v.w - mu) * rs * g4.w + be.w);
    }
    for (int t = 49 + wave; t < 64; t += 4) {
      _Float16* dst = sXn + t * 128 + lane * 4;
      dst[0] = (_Float16)0.f; dst[1] = (_Float16)0.f;
      dst[2] = (_Float16)0.f; dst[3] = (_Float16)0.f;
    }
  }
  __syncthreads();

  // ---- Stage 2: QKV = Xn @ Wqkv^T + b  -> sQKV [64 x 384] ----
  for (int nt = wave * 6; nt < wave * 6 + 6; ++nt) {
    const int coln = nt * 16 + l15;
    const float bias = b_qkv[coln];
    v16h Bf[4];
#pragma unroll
    for (int kt = 0; kt < 4; ++kt)
      Bf[kt] = lwa_load_frag(wqkv + coln * 128 + kt * 32 + 8 * hi);
#pragma unroll
    for (int mt = 0; mt < 4; ++mt) {
      v8f acc = {};
#pragma unroll
      for (int kt = 0; kt < 4; ++kt) {
        const v16h Af = lwa_load_frag(sXn + (mt * 16 + l15) * 128 + kt * 32 + 8 * hi);
        acc = WMMA(Af, Bf[kt], acc);
      }
#pragma unroll
      for (int r = 0; r < 8; ++r)
        sQKV[(mt * 16 + r + 8 * hi) * 384 + coln] = (_Float16)(acc[r] + bias);
    }
  }
  __syncthreads();

  // ---- Stage 3: per-head scores (wave == head), K=32 = one WMMA per tile ----
  const int h = wave;
  float* Sh = sS + h * 4096;
  _Float16* Ph = (_Float16*)Sh;       // probs alias lower half of this head's scores
  {
    const float scale = 0.17677669529663687f;  // 1/sqrt(32)
#pragma unroll
    for (int mt = 0; mt < 4; ++mt) {
      const v16h Qf = lwa_load_frag(sQKV + (mt * 16 + l15) * 384 + h * 32 + 8 * hi);
#pragma unroll
      for (int nt = 0; nt < 4; ++nt) {
        const v16h Kf = lwa_load_frag(sQKV + (nt * 16 + l15) * 384 + 128 + h * 32 + 8 * hi);
        v8f acc = {};
        acc = WMMA(Qf, Kf, acc);
#pragma unroll
        for (int r = 0; r < 8; ++r)
          Sh[(mt * 16 + r + 8 * hi) * 64 + nt * 16 + l15] = acc[r] * scale;
      }
    }
  }

  // ---- Stage 4: masked softmax (49 valid cols); P(f16) written over Sh's lower half.
  // Barriers separate the f32 read phase from the f16 write phase (alias safety).
  for (int rr = 0; rr < 2; ++rr) {
    const int row = lane + 32 * rr;
    const bool active = row < 49;
    float buf[49];
    float mx = -3.0e38f, sum = 0.f;
    if (active) {
#pragma unroll
      for (int j = 0; j < 49; ++j) { buf[j] = Sh[row * 64 + j]; mx = fmaxf(mx, buf[j]); }
#pragma unroll
      for (int j = 0; j < 49; ++j) { buf[j] = __expf(buf[j] - mx); sum += buf[j]; }
    }
    __syncthreads();
    if (active) {
      const float inv = 1.0f / sum;
#pragma unroll
      for (int j = 0; j < 49; ++j) Ph[row * 64 + j] = (_Float16)(buf[j] * inv);
#pragma unroll
      for (int j = 49; j < 64; ++j) Ph[row * 64 + j] = (_Float16)0.f;
    } else if (row < 64) {
#pragma unroll
      for (int j = 0; j < 64; ++j) Ph[row * 64 + j] = (_Float16)0.f;
    }
    __syncthreads();
  }

  // ---- Stage 5a: transpose V_h into the (now free) upper half of this head's
  //       score buffer: sVt[d][n] = V[n][d], 32x64 f16. The softmax barriers
  //       above guarantee all f32 score reads in this byte range are done. ----
  _Float16* sVt = Ph + 4096;   // bytes 8K..12K of the 16 KB head region
  for (int idx = lane; idx < 2048; idx += 32) {
    const int d = idx >> 6, n = idx & 63;
    sVt[idx] = sQKV[n * 384 + 256 + h * 32 + d];
  }

  // ---- Stage 5b: O = P @ V -> sO, head h's 32 columns (B frags now contiguous) ----
#pragma unroll
  for (int mt = 0; mt < 4; ++mt) {
#pragma unroll
    for (int dt = 0; dt < 2; ++dt) {
      v8f acc = {};
#pragma unroll
      for (int kt = 0; kt < 2; ++kt) {
        const v16h Af = lwa_load_frag(Ph  + (mt * 16 + l15) * 64 + kt * 32 + 8 * hi);
        const v16h Bf = lwa_load_frag(sVt + (dt * 16 + l15) * 64 + kt * 32 + 8 * hi);
        acc = WMMA(Af, Bf, acc);
      }
#pragma unroll
      for (int r = 0; r < 8; ++r)
        sO[(mt * 16 + r + 8 * hi) * 128 + h * 32 + dt * 16 + l15] = (_Float16)acc[r];
    }
  }
  __syncthreads();

  // ---- Stage 6: MO = O @ Wmo^T + b_mo -> sXn (reused) ----
  _Float16* sMO = sXn;
  for (int nt = wave * 2; nt < wave * 2 + 2; ++nt) {
    const int coln = nt * 16 + l15;
    const float bias = b_mo[coln];
    v16h Bf[4];
#pragma unroll
    for (int kt = 0; kt < 4; ++kt)
      Bf[kt] = lwa_load_frag(wmo + coln * 128 + kt * 32 + 8 * hi);
#pragma unroll
    for (int mt = 0; mt < 4; ++mt) {
      v8f acc = {};
#pragma unroll
      for (int kt = 0; kt < 4; ++kt) {
        const v16h Af = lwa_load_frag(sO + (mt * 16 + l15) * 128 + kt * 32 + 8 * hi);
        acc = WMMA(Af, Bf[kt], acc);
      }
#pragma unroll
      for (int r = 0; r < 8; ++r)
        sMO[(mt * 16 + r + 8 * hi) * 128 + coln] = (_Float16)(acc[r] + bias);
    }
  }
  __syncthreads();

  // ---- Stage 7: out = x + MO @ Wout^T + b_out (only the 49 valid rows) ----
  for (int nt = wave * 2; nt < wave * 2 + 2; ++nt) {
    const int coln = nt * 16 + l15;
    const float bias = b_out[coln];
    v16h Bf[4];
#pragma unroll
    for (int kt = 0; kt < 4; ++kt)
      Bf[kt] = lwa_load_frag(wout + coln * 128 + kt * 32 + 8 * hi);
#pragma unroll
    for (int mt = 0; mt < 4; ++mt) {
      v8f acc = {};
#pragma unroll
      for (int kt = 0; kt < 4; ++kt) {
        const v16h Af = lwa_load_frag(sMO + (mt * 16 + l15) * 128 + kt * 32 + 8 * hi);
        acc = WMMA(Af, Bf[kt], acc);
      }
#pragma unroll
      for (int r = 0; r < 8; ++r) {
        const int row = mt * 16 + r + 8 * hi;
        if (row < 49) {
          const int tr = row / 7, tc = row - tr * 7;
          const long idx = (gbase + tr * 112 + tc) * 128 + coln;
          out[idx] = x[idx] + acc[r] + bias;
        }
      }
    }
  }
}

extern "C" void kernel_launch(void* const* d_in, const int* in_sizes, int n_in,
                              void* d_out, int out_size, void* d_ws, size_t ws_size,
                              hipStream_t stream) {
  (void)in_sizes; (void)n_in; (void)out_size; (void)ws_size;
  const float* x     = (const float*)d_in[0];
  const float* ln_g  = (const float*)d_in[1];
  const float* ln_b  = (const float*)d_in[2];
  const float* w_qkv = (const float*)d_in[3];
  const float* b_qkv = (const float*)d_in[4];
  const float* w_mo  = (const float*)d_in[5];
  const float* b_mo  = (const float*)d_in[6];
  const float* w_out = (const float*)d_in[7];
  const float* b_out = (const float*)d_in[8];
  // d_in[9]=H, d_in[10]=W are static (112); unused.

  _Float16* wh = (_Float16*)d_ws;  // 384*128 + 2*128*128 f16 = 160 KB
  const int total = 384 * 128 + 2 * 128 * 128;
  lwa_prep_weights<<<(total + 255) / 256, 256, 0, stream>>>(w_qkv, w_mo, w_out, wh);

  lwa_win_attn<<<4096, 128, 0, stream>>>(
      x, ln_g, ln_b, b_qkv, b_mo, b_out,
      wh, wh + 384 * 128, wh + 384 * 128 + 128 * 128,
      (float*)d_out);
}
